// SLXMlp_47304769798199
// MI455X (gfx1250) — matrix operand order repaired and checked by reading, not compiled, not run on hardware
//
#include <hip/hip_runtime.h>
#include <hip/hip_bf16.h>
#include <math.h>

// ---------------------------------------------------------------------------
// SLX-MLP for MI455X (gfx1250), wave32 + v_wmma_f32_16x16x32_bf16.
//
// Factored low-rank path (rank 102 -> padded 128) + stripe-sparse path folded
// into augmented-K WMMA tiles. ~19 GFLOP total, ~200MB traffic -> memory-
// roofline bound (~9us at 23.3 TB/s); everything L2-resident (192MB).
//
// B-slab staging uses the gfx1250 async global->LDS engine
// (GLOBAL_LOAD_ASYNC_TO_LDS_B128 + s_wait_asynccnt): per the compiler's own
// diagnostic the builtin takes (int4 AS1*, int4 AS3*, i32 offset, i32 cpol).
// ---------------------------------------------------------------------------

typedef __bf16 bf16;
typedef bf16  v16bf __attribute__((ext_vector_type(16)));
typedef bf16  v8bf  __attribute__((ext_vector_type(8)));
typedef float v8f   __attribute__((ext_vector_type(8)));
typedef int   v4i   __attribute__((ext_vector_type(4)));

typedef __attribute__((address_space(1))) v4i* gptr_v4i;   // global (__device__)
typedef __attribute__((address_space(3))) v4i* lptr_v4i;   // LDS (__shared__)

#define M_TOK   8192     // 8 * 1024 tokens
#define IN_F    1024
#define HID_F   4096
#define OUT_F   1024
#define RANK    102
#define RPAD    128      // rank padded to multiple of 32
#define KCHUNK  128      // K slab staged in LDS (16 cols x 128 k x 2B = 4KB)

#if defined(__has_builtin)
#if __has_builtin(__builtin_amdgcn_global_load_async_to_lds_b128) && \
    __has_builtin(__builtin_amdgcn_s_wait_asynccnt)
#define HAVE_ASYNC_LDS 1
#endif
#endif
#ifndef HAVE_ASYNC_LDS
#define HAVE_ASYNC_LDS 0
#endif

// ---------------------------------------------------------------------------
// Fragment loaders (wave32; A and B share contiguous k order, k = k0 + 16*(lane/16) + s)
//
// A (16-bit, 16xK tile at k0..k0+31):
//   lanes 0-15 : row = lane,    halves 0-15 = K k0..k0+15
//   lanes16-31 : row = lane-16, halves 0-15 = K k0+16..k0+31
// B (16-bit, Kx16 tile, column n = lane&15):
//   lanes 0-15 : K k0..k0+15 ;  lanes 16-31 : K k0+16..k0+31
// ---------------------------------------------------------------------------
__device__ __forceinline__ v16bf load_a_frag(const bf16* A, int lda, int m0,
                                             int k0, int lane) {
  int row = m0 + (lane & 15);
  int kk  = k0 + ((lane >> 4) << 4);          // +0 or +16
  const bf16* p = A + (size_t)row * lda + kk;
  v8bf lo = *(const v8bf*)(p);
  v8bf hi = *(const v8bf*)(p + 8);
  v16bf r;
#pragma unroll
  for (int i = 0; i < 8; ++i) { r[i] = lo[i]; r[i + 8] = hi[i]; }
  return r;
}

// B fragment from an LDS-staged 16 x KCHUNK tile (row-major, stride KCHUNK).
__device__ __forceinline__ v16bf load_b_frag_lds(const bf16* Bs, int k0, int lane) {
  int col = lane & 15;
  int kk  = k0 + ((lane >> 4) << 4);          // +0 or +16
  return *(const v16bf*)(Bs + col * KCHUNK + kk);   // -> ds_load_b128 x2
}

__device__ __forceinline__ v8f wmma_bf16(v16bf a, v16bf b, v8f c) {
  return __builtin_amdgcn_wmma_f32_16x16x32_bf16(false, a, false, b,
                                                 (short)0, c, false, false);
}

// Cooperative stage of one 16 x KCHUNK slab of Bm (N-major, ldb K-stride)
// into LDS: 256 threads x 16B each. Uses the async global->LDS engine when
// available (ASYNCcnt-tracked, no VGPR round-trip).
__device__ __forceinline__ void stage_b(bf16* Bs, const bf16* Bm, int ldb,
                                        int n0, int kc) {
  int r  = threadIdx.x >> 4;                  // 0..15  (output column)
  int kk = (threadIdx.x & 15) << 3;           // 0..120 step 8
  const bf16* src = Bm + (size_t)(n0 + r) * ldb + kc + kk;
  bf16*       dst = Bs + r * KCHUNK + kk;
#if HAVE_ASYNC_LDS
  __builtin_amdgcn_global_load_async_to_lds_b128(
      (gptr_v4i)(void*)(bf16*)src, (lptr_v4i)(void*)dst, 0, 0);
#else
  *(v8bf*)dst = *(const v8bf*)src;
#endif
}

__device__ __forceinline__ void stage_fence() {
#if HAVE_ASYNC_LDS
  __builtin_amdgcn_s_wait_asynccnt(0);        // own async copies landed in LDS
#endif
  __syncthreads();                            // visible to all waves
}

// ---------------------------------------------------------------------------
// Kernel: fp32 -> bf16 with 2-D zero padding (used for x and U/V factors).
// ---------------------------------------------------------------------------
__global__ void pad2d_f32_to_bf16(const float* __restrict__ src,
                                  bf16* __restrict__ dst,
                                  int dst_rows, int dst_cols,
                                  int src_rows, int src_cols) {
  int idx = blockIdx.x * blockDim.x + threadIdx.x;
  int total = dst_rows * dst_cols;
  if (idx >= total) return;
  int r = idx / dst_cols, c = idx % dst_cols;
  float v = (r < src_rows && c < src_cols) ? src[(size_t)r * src_cols + c] : 0.0f;
  dst[idx] = (bf16)v;
}

// ---------------------------------------------------------------------------
// Kernel: D[m,n] = sum_k A[m,k] * Bm[n,k]  (both K-contiguous, bf16 in/out).
// Block = 8 waves sharing one 16-wide n-tile; each wave owns MI m-subtiles
// (16 rows each) so the LDS-resident B fragment is reused MI times from regs.
// ---------------------------------------------------------------------------
template <int MI>
__global__ void gemm_bf16_nt(const bf16* __restrict__ A,
                             const bf16* __restrict__ Bm,
                             bf16* __restrict__ D,
                             int K, int lda, int ldb, int ldd) {
  __shared__ bf16 Bs[16 * KCHUNK];            // 4KB slab
  int lane = threadIdx.x & 31;
  int wave = threadIdx.x >> 5;
  int m0 = blockIdx.x * (128 * MI) + wave * (16 * MI);
  int n0 = blockIdx.y * 16;

  v8f acc[MI];
#pragma unroll
  for (int mi = 0; mi < MI; ++mi) acc[mi] = (v8f){};

  for (int kc = 0; kc < K; kc += KCHUNK) {
    __syncthreads();                          // protect prior slab reads
    stage_b(Bs, Bm, ldb, n0, kc);
    stage_fence();
#pragma unroll
    for (int k0 = 0; k0 < KCHUNK; k0 += 32) {
      v16bf b = load_b_frag_lds(Bs, k0, lane);
#pragma unroll
      for (int mi = 0; mi < MI; ++mi) {
        v16bf a = load_a_frag(A, lda, m0 + 16 * mi, kc + k0, lane);
        acc[mi] = wmma_bf16(a, b, acc[mi]);
      }
    }
  }

  int col = n0 + (lane & 15);
#pragma unroll
  for (int mi = 0; mi < MI; ++mi) {
    int rbase = m0 + 16 * mi + ((lane >> 4) << 3);
#pragma unroll
    for (int v = 0; v < 8; ++v)
      D[(size_t)(rbase + v) * ldd + col] = (bf16)acc[mi][v];
  }
}

// ---------------------------------------------------------------------------
// Layer 1: h = gelu(xr1 @ U1^T + stripe_sparse(x, Ws1) + b1)  -> bf16
// Sparse stripes (3 stripes x 3 taps, centers o>>2, offsets {0,341,682}) are
// folded into ONE augmented K=32 WMMA per tile: for a 16-wide o-tile the taps
// touch 3 runs of 6 consecutive x columns -> 18 augmented K slots.
// ---------------------------------------------------------------------------
__global__ void slx_layer1(const bf16* __restrict__ xr,
                           const bf16* __restrict__ U1b,
                           const bf16* __restrict__ xb,
                           const float* __restrict__ Ws1,
                           const float* __restrict__ b1,
                           bf16* __restrict__ hb) {
  __shared__ bf16 Bs[16 * KCHUNK];
  int lane = threadIdx.x & 31;
  int wave = threadIdx.x >> 5;
  int m0 = blockIdx.x * 512 + wave * 64;      // MI = 4
  int n0 = blockIdx.y * 16;

  v8f acc[4];
#pragma unroll
  for (int mi = 0; mi < 4; ++mi) acc[mi] = (v8f){};

  // K = RPAD = one LDS slab of U1 rows n0..n0+15
  stage_b(Bs, U1b, RPAD, n0, 0);
  stage_fence();
#pragma unroll
  for (int k0 = 0; k0 < RPAD; k0 += 32) {
    v16bf b = load_b_frag_lds(Bs, k0, lane);
#pragma unroll
    for (int mi = 0; mi < 4; ++mi) {
      v16bf a = load_a_frag(xr, RPAD, m0 + 16 * mi, k0, lane);
      acc[mi] = wmma_bf16(a, b, acc[mi]);
    }
  }

  // ---- augmented sparse tile: K slots j in [0,18) ; j = s*6 + d
  const int OFF0 = 0, OFF1 = 341, OFF2 = 682;   // (k*IN_F)/3
  int base = n0 >> 2;                            // center of o = n0

  v16bf bb;                                      // B_aug[j][o] = Ws1[o,col_j] iff tap
  {
    int o    = n0 + (lane & 15);
    int rel0 = (o >> 2) - base;                  // 0..3
    int jb   = (lane >> 4) << 4;
#pragma unroll
    for (int i = 0; i < 16; ++i) {
      int j = jb + i;
      float v = 0.0f;
      if (j < 18) {
        int s = j / 6, d = j - 6 * s;
        int t = d - rel0;                        // tap index
        if (t >= 0 && t < 3) {
          int off = (s == 0) ? OFF0 : (s == 1) ? OFF1 : OFF2;
          int colx = (base + off + d) & (IN_F - 1);
          v = Ws1[(size_t)o * IN_F + colx];
        }
      }
      bb[i] = (bf16)v;
    }
  }
#pragma unroll
  for (int mi = 0; mi < 4; ++mi) {
    v16bf aa;                                    // A_aug[m][j] = x[m, base+off_s+d]
    int row = m0 + 16 * mi + (lane & 15);
    int jb0 = (lane >> 4) << 4;
#pragma unroll
    for (int i = 0; i < 16; ++i) {
      int j = jb0 + i;
      bf16 v = (bf16)0.0f;
      if (j < 18) {
        int s = j / 6, d = j - 6 * s;
        int off = (s == 0) ? OFF0 : (s == 1) ? OFF1 : OFF2;
        int colx = (base + off + d) & (IN_F - 1);
        v = xb[(size_t)row * IN_F + colx];
      }
      aa[i] = v;
    }
    acc[mi] = wmma_bf16(aa, bb, acc[mi]);
  }

  // ---- epilogue: bias + exact-erf GELU, store bf16
  int col = n0 + (lane & 15);
  float bias = b1[col];
#pragma unroll
  for (int mi = 0; mi < 4; ++mi) {
    int rbase = m0 + 16 * mi + ((lane >> 4) << 3);
#pragma unroll
    for (int v = 0; v < 8; ++v) {
      float h = acc[mi][v] + bias;
      h = 0.5f * h * (1.0f + erff(h * 0.70710678118654752f));
      hb[(size_t)(rbase + v) * HID_F + col] = (bf16)h;
    }
  }
}

// ---------------------------------------------------------------------------
// Layer 2: out = xr2 @ U2^T + stripe_sparse(h, Ws2) + b2  -> fp32
// Single stripe, center = 4*o: a 16-wide o-tile touches 64 CONTIGUOUS h
// columns -> two augmented K=32 WMMAs with plain contiguous A loads.
// ---------------------------------------------------------------------------
__global__ void slx_layer2(const bf16* __restrict__ xr2,
                           const bf16* __restrict__ U2b,
                           const bf16* __restrict__ hb,
                           const float* __restrict__ Ws2,
                           const float* __restrict__ b2,
                           float* __restrict__ out) {
  __shared__ bf16 Bs[16 * KCHUNK];
  int lane = threadIdx.x & 31;
  int wave = threadIdx.x >> 5;
  int m0 = blockIdx.x * 512 + wave * 64;      // MI = 4
  int n0 = blockIdx.y * 16;

  v8f acc[4];
#pragma unroll
  for (int mi = 0; mi < 4; ++mi) acc[mi] = (v8f){};

  stage_b(Bs, U2b, RPAD, n0, 0);
  stage_fence();
#pragma unroll
  for (int k0 = 0; k0 < RPAD; k0 += 32) {
    v16bf b = load_b_frag_lds(Bs, k0, lane);
#pragma unroll
    for (int mi = 0; mi < 4; ++mi) {
      v16bf a = load_a_frag(xr2, RPAD, m0 + 16 * mi, k0, lane);
      acc[mi] = wmma_bf16(a, b, acc[mi]);
    }
  }

#pragma unroll
  for (int part = 0; part < 2; ++part) {
    int jbase = part * 32;                       // jj in [0,64): col = 4*n0 + jj
    v16bf bb;
    {
      int o  = n0 + (lane & 15);
      int jb = (lane >> 4) << 4;
#pragma unroll
      for (int i = 0; i < 16; ++i) {
        int jj = jbase + jb + i;
        int t  = jj - 4 * (o - n0);              // tap index
        float v = (t >= 0 && t < 3) ? Ws2[(size_t)o * HID_F + 4 * n0 + jj] : 0.0f;
        bb[i] = (bf16)v;
      }
    }
#pragma unroll
    for (int mi = 0; mi < 4; ++mi) {
      v16bf aa;
      int row = m0 + 16 * mi + (lane & 15);
      int jb0 = (lane >> 4) << 4;
#pragma unroll
      for (int i = 0; i < 16; ++i) {
        int j  = jb0 + i;
        int jj = jbase + j;
        aa[i] = hb[(size_t)row * HID_F + 4 * n0 + jj];
      }
      acc[mi] = wmma_bf16(aa, bb, acc[mi]);
    }
  }

  int col = n0 + (lane & 15);
  float bias = b2[col];
#pragma unroll
  for (int mi = 0; mi < 4; ++mi) {
    int rbase = m0 + 16 * mi + ((lane >> 4) << 3);
#pragma unroll
    for (int v = 0; v < 8; ++v)
      out[(size_t)(rbase + v) * OUT_F + col] = acc[mi][v] + bias;
  }
}

// ---------------------------------------------------------------------------
// Host-side launch
// ---------------------------------------------------------------------------
static inline size_t align256(size_t v) { return (v + 255) & ~(size_t)255; }

extern "C" void kernel_launch(void* const* d_in, const int* in_sizes, int n_in,
                              void* d_out, int out_size, void* d_ws, size_t ws_size,
                              hipStream_t stream) {
  (void)in_sizes; (void)n_in; (void)out_size; (void)ws_size;

  const float* x   = (const float*)d_in[0];
  const float* U1  = (const float*)d_in[1];
  const float* V1  = (const float*)d_in[2];
  const float* Ws1 = (const float*)d_in[3];
  const float* b1  = (const float*)d_in[4];
  const float* U2  = (const float*)d_in[5];
  const float* V2  = (const float*)d_in[6];
  const float* Ws2 = (const float*)d_in[7];
  const float* b2  = (const float*)d_in[8];
  // masks d_in[9], d_in[10] are analytic -> recomputed on the fly

  // ---- workspace carve (all 256B-aligned)
  char*  w = (char*)d_ws;
  size_t off = 0;
  bf16* xb   = (bf16*)(w + off); off = align256(off + (size_t)M_TOK * IN_F  * 2);
  bf16* hb   = (bf16*)(w + off); off = align256(off + (size_t)M_TOK * HID_F * 2);
  bf16* xr1b = (bf16*)(w + off); off = align256(off + (size_t)M_TOK * RPAD  * 2);
  bf16* xr2b = (bf16*)(w + off); off = align256(off + (size_t)M_TOK * RPAD  * 2);
  bf16* V1b  = (bf16*)(w + off); off = align256(off + (size_t)RPAD  * IN_F  * 2);
  bf16* U1b  = (bf16*)(w + off); off = align256(off + (size_t)HID_F * RPAD  * 2);
  bf16* V2b  = (bf16*)(w + off); off = align256(off + (size_t)RPAD  * HID_F * 2);
  bf16* U2b  = (bf16*)(w + off); off = align256(off + (size_t)OUT_F * RPAD  * 2);

  const dim3 blk(256);
  auto g1 = [](int total) { return dim3((unsigned)((total + 255) / 256)); };

  // ---- pack / convert (padding rewritten with zeros every call: deterministic)
  pad2d_f32_to_bf16<<<g1(M_TOK * IN_F), blk, 0, stream>>>(x,  xb,  M_TOK, IN_F,  M_TOK, IN_F);
  pad2d_f32_to_bf16<<<g1(RPAD * IN_F),  blk, 0, stream>>>(V1, V1b, RPAD,  IN_F,  RANK,  IN_F);
  pad2d_f32_to_bf16<<<g1(HID_F * RPAD), blk, 0, stream>>>(U1, U1b, HID_F, RPAD,  HID_F, RANK);
  pad2d_f32_to_bf16<<<g1(RPAD * HID_F), blk, 0, stream>>>(V2, V2b, RPAD,  HID_F, RANK,  HID_F);
  pad2d_f32_to_bf16<<<g1(OUT_F * RPAD), blk, 0, stream>>>(U2, U2b, OUT_F, RPAD,  OUT_F, RANK);

  // ---- xr1 = x @ V1^T        (8192 x 128, K=1024)   MI=1 -> 512 blocks
  gemm_bf16_nt<1><<<dim3(M_TOK / 128, RPAD / 16), blk, 0, stream>>>(
      xb, V1b, xr1b, IN_F, IN_F, IN_F, RPAD);

  // ---- h = gelu(xr1 @ U1^T + sparse + b1)   (8192 x 4096)
  slx_layer1<<<dim3(M_TOK / 512, HID_F / 16), blk, 0, stream>>>(
      xr1b, U1b, xb, Ws1, b1, hb);

  // ---- xr2 = h @ V2^T        (8192 x 128, K=4096)   MI=1 -> 512 blocks
  gemm_bf16_nt<1><<<dim3(M_TOK / 128, RPAD / 16), blk, 0, stream>>>(
      hb, V2b, xr2b, HID_F, HID_F, HID_F, RPAD);

  // ---- out = xr2 @ U2^T + sparse + b2       (8192 x 1024, fp32)
  slx_layer2<<<dim3(M_TOK / 512, OUT_F / 16), blk, 0, stream>>>(
      xr2b, U2b, hb, Ws2, b2, (float*)d_out);
}
